// TraditionalSpikingModel_35510789604047
// MI455X (gfx1250) — matrix operand order repaired and checked by reading, not compiled, not run on hardware
//
#include <hip/hip_runtime.h>
#include <hip/hip_bf16.h>

// ---------------------------------------------------------------------------
// Spiking net forward for MI455X (gfx1250, wave32, WMMA).
//
// Precision: fsq() outputs are exact integers in {0..7} -> exact in FP16.
// Weights split W = f16(W) + f16(W - f16(W)); two v_wmma_f32_16x16x32_f16 per
// K-step give ~fp32 accuracy at FP16-WMMA rate.
//
// GEMM data path: B(hi/lo) tile double-buffered through LDS (shared by all 8
// waves -> 8x less global B traffic). Per 64-wide K stage, ALL 32 fragment
// ds_loads are issued ahead of the 16-WMMA burst (enforced with
// sched_group_barrier), so the in-order DS counter covers the burst with one
// partial wait instead of one wait per WMMA. A fragments register-pipelined.
// ---------------------------------------------------------------------------

typedef __attribute__((ext_vector_type(16))) _Float16 v16h;
typedef __attribute__((ext_vector_type(8)))  _Float16 v8h;
typedef __attribute__((ext_vector_type(8)))  float    v8f;

namespace cfg {
constexpr int T   = 500;
constexpr int B   = 64;
constexpr int IN  = 440;
constexpr int INP = 448;   // IN padded to multiple of 64 (K stage)
constexpr int H   = 1024;
constexpr int L   = 4;
constexpr int OUT = 1944;
constexpr int OUTP = 1984; // OUT padded to multiple of 64 (N tile)
constexpr int M   = T * B; // 32000 = 250 * 128 (exact)
constexpr int BPITCH = 72; // LDS row pitch (halfs): 144B -> conflict-free b128
constexpr float BN_SCALE = 0.99999500003749975f; // 1/sqrt(1 + 1e-5)
}

static __device__ __forceinline__ v16h cat8(v8h a, v8h b) {
  return __builtin_shufflevector(a, b, 0, 1, 2, 3, 4, 5, 6, 7,
                                 8, 9, 10, 11, 12, 13, 14, 15);
}

// ---------------------------------------------------------------------------
// Weight prep: fp32 [N,K] -> f16 hi/lo pair, padded to [?,Kpad] (zero fill).
// ---------------------------------------------------------------------------
__global__ void snn_prep_weights(const float* __restrict__ W,
                                 _Float16* __restrict__ hi,
                                 _Float16* __restrict__ lo,
                                 int N, int K, int Kpad, int total) {
  int tid = blockIdx.x * blockDim.x + threadIdx.x;
  if (tid >= total) return;
  int n = tid / Kpad, k = tid - n * Kpad;
  float v = (n < N && k < K) ? W[(size_t)n * K + k] : 0.0f;
  _Float16 h = (_Float16)v;
  float r = v - (float)h;
  hi[tid] = h;
  lo[tid] = (_Float16)r;
}

// ---------------------------------------------------------------------------
// Input FSQ scan: sequential over T per (b, i) chain; outputs exact {0..7}
// as f16 into padded activation buffer [M, INP].
// ---------------------------------------------------------------------------
__global__ void snn_input_fsq(const float* __restrict__ xs,
                              _Float16* __restrict__ out) {
  int tid = blockIdx.x * blockDim.x + threadIdx.x;
  if (tid >= cfg::B * cfg::INP) return;
  int b = tid / cfg::INP, i = tid - b * cfg::INP;
  if (i >= cfg::IN) { // zero padding columns every call (buffer is poisoned)
    _Float16* op = out + (size_t)b * cfg::INP + i;
    for (int t = 0; t < cfg::T; ++t) op[(size_t)t * cfg::B * cfg::INP] = (_Float16)0.0f;
    return;
  }
  const float* xp = xs + (size_t)b * cfg::IN + i;
  _Float16* op = out + (size_t)b * cfg::INP + i;
  float vmem = 0.0f;
  for (int t = 0; t < cfg::T; ++t) {
    vmem += xp[(size_t)t * cfg::B * cfg::IN];
    float q = fminf(fmaxf(rintf(vmem), 0.0f), 7.0f); // round-to-even like jnp.round
    vmem -= q;
    op[(size_t)t * cfg::B * cfg::INP] = (_Float16)q;
  }
}

// ---------------------------------------------------------------------------
// LIF scan: sequential over T per (b, h) chain. Reads BN'd f32 projections,
// writes spike outputs (exact {0..7}) as f16 for the next GEMM.
// ---------------------------------------------------------------------------
__global__ void snn_lif_scan(const float* __restrict__ ws,
                             const float* __restrict__ tau,
                             _Float16* __restrict__ out) {
  int tid = blockIdx.x * blockDim.x + threadIdx.x;
  const int BH = cfg::B * cfg::H;
  if (tid >= BH) return;
  float tv = tau[tid & (cfg::H - 1)];
  float syn = 0.0f, vmem = 0.0f;
  for (int t = 0; t < cfg::T; ++t) {
    float w = ws[(size_t)t * BH + tid];
    syn = w + tv * syn;
    vmem += syn;
    float q = fminf(fmaxf(rintf(vmem), 0.0f), 7.0f);
    vmem -= q;
    out[(size_t)t * BH + tid] = (_Float16)q;
  }
}

// ---------------------------------------------------------------------------
// WMMA GEMM + fused BatchNorm(eval) epilogue.
//   C[m,n] = BN( sum_k A[m,k] * (Bhi[n,k] + Blo[n,k]) )
// A: f16 [M, Kpad] row-major; Bhi/Blo: f16 [Npad, Kpad] row-major (K contig).
// Block = 8 waves; wave computes 16x64; block tile 128(M) x 64(N) x 64(Kstage).
// Lane layouts (A/B: contiguous K, this code's own order; C per CDNA5 ISA §7.12.2):
//   A 16x32: lane L holds row M=L&15, K range 16*(L>>4) .. +15 (contig).
//   B 32x16: lane L holds col N=L&15, K range 16*(L>>4) .. +15 (contig).
//   C 16x16: element (vgpr r, lane L) = (M = r + 8*(L>>4), N = L&15).
// ---------------------------------------------------------------------------
__global__ __launch_bounds__(256) void snn_gemm_bn(
    const _Float16* __restrict__ A,
    const _Float16* __restrict__ Bhi,
    const _Float16* __restrict__ Blo,
    float* __restrict__ C,
    const float* __restrict__ gamma,
    const float* __restrict__ beta,
    int Kpad, int Nact, int ldC) {
  using namespace cfg;
  // [buf][hi/lo][row(n)][k] with padded pitch for conflict-free ds_load_b128.
  __shared__ _Float16 lds[2][2][64][BPITCH];

  const int lane = threadIdx.x & 31;
  const int wave = threadIdx.x >> 5;
  const int ll = lane & 15;   // row (A) / col (B,C) within tile
  const int lh = lane >> 4;   // lane-half: selects K/M sub-chunk
  const int Mbase = blockIdx.y * 128 + wave * 16;
  const int Nbase = blockIdx.x * 64;

  // Staging role: thread -> (row 0..63, 16-half chunk 0..3) of the B stage.
  const int srow   = threadIdx.x >> 2;
  const int schunk = (threadIdx.x & 3) * 16;
  const _Float16* gHi = Bhi + (size_t)(Nbase + srow) * Kpad + schunk;
  const _Float16* gLo = Blo + (size_t)(Nbase + srow) * Kpad + schunk;

  v8f acc[4] = {{0.f,0.f,0.f,0.f,0.f,0.f,0.f,0.f},
                {0.f,0.f,0.f,0.f,0.f,0.f,0.f,0.f},
                {0.f,0.f,0.f,0.f,0.f,0.f,0.f,0.f},
                {0.f,0.f,0.f,0.f,0.f,0.f,0.f,0.f}};

  const _Float16* aRow = A + (size_t)(Mbase + ll) * Kpad + 16 * lh;

  // ---- prologue: stage-0 B in registers, K-step-0 A in registers ----
  v8h h0 = *(const v8h*)(gHi);
  v8h h1 = *(const v8h*)(gHi + 8);
  v8h l0 = *(const v8h*)(gLo);
  v8h l1 = *(const v8h*)(gLo + 8);
  v8h aLo = *(const v8h*)(aRow);
  v8h aHi = *(const v8h*)(aRow + 8);

  const int nstages = Kpad >> 6; // 64-wide K stages
  for (int st = 0; st < nstages; ++st) {
    const int buf = st & 1;
    const int k0 = st << 6;

    // Publish stage st to LDS (regs were loaded last iteration / prologue).
    __syncthreads(); // waves done reading this buffer two stages ago
    *(v8h*)&lds[buf][0][srow][schunk]     = h0;
    *(v8h*)&lds[buf][0][srow][schunk + 8] = h1;
    *(v8h*)&lds[buf][1][srow][schunk]     = l0;
    *(v8h*)&lds[buf][1][srow][schunk + 8] = l1;
    __syncthreads();

    // Kick off stage st+1 global loads; they overlap the WMMA burst below.
    if (st + 1 < nstages) {
      h0 = *(const v8h*)(gHi + k0 + 64);
      h1 = *(const v8h*)(gHi + k0 + 72);
      l0 = *(const v8h*)(gLo + k0 + 64);
      l1 = *(const v8h*)(gLo + k0 + 72);
      if (st + 2 < nstages) {
        __builtin_prefetch(gHi + k0 + 128, 0, 1); // global_prefetch_b8
        __builtin_prefetch(gLo + k0 + 128, 0, 1);
      }
    }

    // A fragments: K-step 0 comes from the pipeline regs; load K-step 1 now,
    // and refill the pipeline regs with next stage's K-step 0.
    v16h a0 = cat8(aLo, aHi);
    v8h t0 = *(const v8h*)(aRow + k0 + 32);
    v8h t1 = *(const v8h*)(aRow + k0 + 40);
    v16h a1 = cat8(t0, t1);
    if (st + 1 < nstages) {
      aLo = *(const v8h*)(aRow + k0 + 64);
      aHi = *(const v8h*)(aRow + k0 + 72);
    }

    // ALL 32 fragment ds_loads for both K-steps, in distinct SSA values,
    // ahead of the 16-WMMA burst.
    v16h bh0[4], bl0[4], bh1[4], bl1[4];
#pragma unroll
    for (int j = 0; j < 4; ++j) {
      const _Float16* ph = &lds[buf][0][j * 16 + ll][16 * lh];
      const _Float16* pl = &lds[buf][1][j * 16 + ll][16 * lh];
      bh0[j] = cat8(*(const v8h*)(ph),      *(const v8h*)(ph + 8));
      bl0[j] = cat8(*(const v8h*)(pl),      *(const v8h*)(pl + 8));
      bh1[j] = cat8(*(const v8h*)(ph + 32), *(const v8h*)(ph + 40));
      bl1[j] = cat8(*(const v8h*)(pl + 32), *(const v8h*)(pl + 40));
    }

#pragma unroll
    for (int j = 0; j < 4; ++j) {
      acc[j] = __builtin_amdgcn_wmma_f32_16x16x32_f16(
          false, a0, false, bh0[j], (short)0, acc[j], false, false);
      acc[j] = __builtin_amdgcn_wmma_f32_16x16x32_f16(
          false, a0, false, bl0[j], (short)0, acc[j], false, false);
    }
#pragma unroll
    for (int j = 0; j < 4; ++j) {
      acc[j] = __builtin_amdgcn_wmma_f32_16x16x32_f16(
          false, a1, false, bh1[j], (short)0, acc[j], false, false);
      acc[j] = __builtin_amdgcn_wmma_f32_16x16x32_f16(
          false, a1, false, bl1[j], (short)0, acc[j], false, false);
    }

#if __has_builtin(__builtin_amdgcn_sched_group_barrier)
    // Pin the per-stage schedule: next-stage VMEM reads first, then the 32
    // fragment DS reads, then the 16-WMMA burst. Stops the scheduler from
    // re-interleaving ds_load -> wait -> wmma per fragment (round-2 pathology).
    __builtin_amdgcn_sched_group_barrier(0x020, 8, 0);  // VMEM reads (B st+1, A)
    __builtin_amdgcn_sched_group_barrier(0x100, 32, 0); // DS reads (fragments)
    __builtin_amdgcn_sched_group_barrier(0x008, 16, 0); // WMMA burst
#endif
  }

  // Epilogue: BN(eval, running stats mean=0,var=1): y = x*gamma/sqrt(1+eps)+beta
#pragma unroll
  for (int j = 0; j < 4; ++j) {
    int n = Nbase + j * 16 + ll;
    if (n < Nact) {
      float sc = gamma[n] * BN_SCALE;
      float sh = beta[n];
#pragma unroll
      for (int r = 0; r < 8; ++r) {
        int m = Mbase + r + 8 * lh;
        C[(size_t)m * ldC + n] = acc[j][r] * sc + sh;
      }
    }
  }
}

// ---------------------------------------------------------------------------
// In-place row-wise log-softmax; one 256-thread block per row, LDS reduce.
// ---------------------------------------------------------------------------
__global__ __launch_bounds__(256) void snn_log_softmax(float* __restrict__ x, int N) {
  __shared__ float red[256];
  float* row = x + (size_t)blockIdx.x * N;
  const int t = threadIdx.x;

  float mx = -3.4e38f;
  for (int i = t; i < N; i += 256) mx = fmaxf(mx, row[i]);
  red[t] = mx;
  __syncthreads();
  for (int s = 128; s > 0; s >>= 1) {
    if (t < s) red[t] = fmaxf(red[t], red[t + s]);
    __syncthreads();
  }
  mx = red[0];
  __syncthreads();

  float sum = 0.0f;
  for (int i = t; i < N; i += 256) sum += expf(row[i] - mx);
  red[t] = sum;
  __syncthreads();
  for (int s = 128; s > 0; s >>= 1) {
    if (t < s) red[t] += red[t + s];
    __syncthreads();
  }
  float lse = mx + logf(red[0]);

  for (int i = t; i < N; i += 256) row[i] = row[i] - lse;
}

// ---------------------------------------------------------------------------
// Launch sequence (all on `stream`; stream order = dependency order).
// ---------------------------------------------------------------------------
extern "C" void kernel_launch(void* const* d_in, const int* in_sizes, int n_in,
                              void* d_out, int out_size, void* d_ws, size_t ws_size,
                              hipStream_t stream) {
  (void)in_sizes; (void)n_in; (void)out_size; (void)ws_size;
  using namespace cfg;

  const float* xs       = (const float*)d_in[0]; // [T,B,IN]
  const float* W0       = (const float*)d_in[1]; // [H,IN]
  const float* Ws       = (const float*)d_in[2]; // [L-1,H,H]
  const float* taus     = (const float*)d_in[3]; // [L,H]
  const float* bn_gamma = (const float*)d_in[4]; // [L,H]
  const float* bn_beta  = (const float*)d_in[5]; // [L,H]
  const float* Wf       = (const float*)d_in[6]; // [OUT,H]
  const float* fbn_g    = (const float*)d_in[7]; // [OUT]
  const float* fbn_b    = (const float*)d_in[8]; // [OUT]
  float* logits         = (float*)d_out;         // [M,OUT]

  // ---- scratch carve-up (256B aligned) ----
  char* base = (char*)d_ws;
  size_t off = 0;
  auto alloc = [&](size_t bytes) -> char* {
    char* p = base + off;
    off = (off + bytes + 255) & ~(size_t)255;
    return p;
  };
  _Float16* act0  = (_Float16*)alloc((size_t)M * INP * 2);   // 28.7 MB
  _Float16* actA  = (_Float16*)alloc((size_t)M * H * 2);     // 65.5 MB
  _Float16* actB  = (_Float16*)alloc((size_t)M * H * 2);     // 65.5 MB
  float*    wsbuf = (float*)   alloc((size_t)M * H * 4);     // 131 MB
  _Float16* w0hi  = (_Float16*)alloc((size_t)H * INP * 2);
  _Float16* w0lo  = (_Float16*)alloc((size_t)H * INP * 2);
  _Float16* whhi[3]; _Float16* whlo[3];
  for (int l = 0; l < 3; ++l) {
    whhi[l] = (_Float16*)alloc((size_t)H * H * 2);
    whlo[l] = (_Float16*)alloc((size_t)H * H * 2);
  }
  _Float16* wfhi  = (_Float16*)alloc((size_t)OUTP * H * 2);
  _Float16* wflo  = (_Float16*)alloc((size_t)OUTP * H * 2);

  // ---- weight hi/lo splits ----
  {
    int tot = H * INP;
    snn_prep_weights<<<(tot + 255) / 256, 256, 0, stream>>>(W0, w0hi, w0lo, H, IN, INP, tot);
  }
  for (int l = 0; l < 3; ++l) {
    int tot = H * H;
    snn_prep_weights<<<(tot + 255) / 256, 256, 0, stream>>>(
        Ws + (size_t)l * H * H, whhi[l], whlo[l], H, H, H, tot);
  }
  {
    int tot = OUTP * H;
    snn_prep_weights<<<(tot + 255) / 256, 256, 0, stream>>>(Wf, wfhi, wflo, OUT, H, H, tot);
  }

  // ---- input spike quantization scan ----
  {
    int tot = B * INP;
    snn_input_fsq<<<(tot + 255) / 256, 256, 0, stream>>>(xs, act0);
  }

  // ---- layer 0: GEMM(K=448) + BN -> LIF ----
  snn_gemm_bn<<<dim3(H / 64, M / 128), 256, 0, stream>>>(
      act0, w0hi, w0lo, wsbuf, bn_gamma, bn_beta, INP, H, H);
  snn_lif_scan<<<(B * H) / 256, 256, 0, stream>>>(wsbuf, taus, actA);

  // ---- layers 1..3: GEMM(K=1024) + BN -> LIF (ping-pong) ----
  _Float16* cur = actA;
  _Float16* nxt = actB;
  for (int l = 1; l < L; ++l) {
    snn_gemm_bn<<<dim3(H / 64, M / 128), 256, 0, stream>>>(
        cur, whhi[l - 1], whlo[l - 1], wsbuf,
        bn_gamma + (size_t)l * H, bn_beta + (size_t)l * H, H, H, H);
    snn_lif_scan<<<(B * H) / 256, 256, 0, stream>>>(wsbuf, taus + (size_t)l * H, nxt);
    _Float16* t = cur; cur = nxt; nxt = t;
  }

  // ---- final linear + BN straight into d_out, then in-place log-softmax ----
  snn_gemm_bn<<<dim3(OUTP / 64, M / 128), 256, 0, stream>>>(
      cur, wfhi, wflo, logits, fbn_g, fbn_b, H, OUT, OUT);
  snn_log_softmax<<<M, 256, 0, stream>>>(logits, OUT);
}
